// MSA_Row_Att_34660386079044
// MI455X (gfx1250) — compile-verified, matrix operand order checked by reading, not run on hardware
//
#include <hip/hip_runtime.h>
#include <hip/hip_bf16.h>
#include <math.h>

#define MSA_DIM 256
#define PAIR_DIM 128
#define NH 8
#define HD 32
#define NS 8
#define LL 1024
#define SL (NS * LL)          // 8192
#define LN_EPS 1e-5f
#define QK_SCALE 0.17677669529663687f  // 1/sqrt(32)

typedef __attribute__((ext_vector_type(16))) __bf16 v16bf;
typedef __attribute__((ext_vector_type(8)))  float  v8f;

union FragAB { v16bf v; uint4 q[2]; };

__device__ __forceinline__ unsigned short f2bf(float f) {
  unsigned u = __float_as_uint(f);
  u += 0x7FFFu + ((u >> 16) & 1u);       // round-to-nearest-even
  return (unsigned short)(u >> 16);
}

__device__ __forceinline__ float wave_sum32(float v) {
#pragma unroll
  for (int m = 16; m > 0; m >>= 1) v += __shfl_xor(v, m, 32);
  return v;
}
__device__ __forceinline__ float half_sum16(float v) {   // reduce within 16-lane halves
#pragma unroll
  for (int m = 8; m > 0; m >>= 1) v += __shfl_xor(v, m, 32);
  return v;
}
__device__ __forceinline__ float half_max16(float v) {
#pragma unroll
  for (int m = 8; m > 0; m >>= 1) v = fmaxf(v, __shfl_xor(v, m, 32));
  return v;
}

// ---------------- Kernel 0: LayerNorm(msa) -> bf16 [8192,256] ----------------
__global__ void k_ln_msa(const float* __restrict__ msa,
                         const float* __restrict__ g,
                         const float* __restrict__ b,
                         unsigned short* __restrict__ m_bf) {
  int wave = threadIdx.x >> 5, lane = threadIdx.x & 31;
  int row = blockIdx.x * 8 + wave;
  const float4* src = (const float4*)(msa + (size_t)row * MSA_DIM);
  float4 x0 = src[lane];
  float4 x1 = src[lane + 32];
  float s  = x0.x + x0.y + x0.z + x0.w + x1.x + x1.y + x1.z + x1.w;
  float sq = x0.x*x0.x + x0.y*x0.y + x0.z*x0.z + x0.w*x0.w
           + x1.x*x1.x + x1.y*x1.y + x1.z*x1.z + x1.w*x1.w;
  s = wave_sum32(s);  sq = wave_sum32(sq);
  float mean = s * (1.0f / 256.0f);
  float var  = sq * (1.0f / 256.0f) - mean * mean;
  float rstd = rsqrtf(var + LN_EPS);
  int d0 = lane * 4, d1 = 128 + lane * 4;
  float a0[4] = {x0.x, x0.y, x0.z, x0.w};
  float a1[4] = {x1.x, x1.y, x1.z, x1.w};
  ushort4 o0, o1;
  unsigned short* p0 = (unsigned short*)&o0;
  unsigned short* p1 = (unsigned short*)&o1;
#pragma unroll
  for (int q = 0; q < 4; ++q) {
    p0[q] = f2bf((a0[q] - mean) * rstd * g[d0 + q] + b[d0 + q]);
    p1[q] = f2bf((a1[q] - mean) * rstd * g[d1 + q] + b[d1 + q]);
  }
  *(ushort4*)(m_bf + (size_t)row * 256 + d0) = o0;
  *(ushort4*)(m_bf + (size_t)row * 256 + d1) = o1;
}

// ------------- Kernel 1: convert+transpose weights to bf16 -------------------
// W4t[w][n][k] = W_w[k][n] for w in {q,k,v,g};  Wot[n][k] = Wo[k][n]
__global__ void k_wconv(const float* __restrict__ Wq, const float* __restrict__ Wk,
                        const float* __restrict__ Wv, const float* __restrict__ Wg,
                        const float* __restrict__ Wo,
                        unsigned short* __restrict__ W4t,
                        unsigned short* __restrict__ Wot) {
  int tid = blockIdx.x * 256 + threadIdx.x;      // < 5*65536
  int w = tid >> 16, rem = tid & 65535;
  int n = rem >> 8, k = rem & 255;
  const float* src = (w == 0) ? Wq : (w == 1) ? Wk : (w == 2) ? Wv : (w == 3) ? Wg : Wo;
  unsigned short h = f2bf(src[k * 256 + n]);
  if (w < 4) W4t[(size_t)w * 65536 + n * 256 + k] = h;
  else       Wot[(size_t)n * 256 + k] = h;
}

// -------- Kernel 2: LayerNorm(pair) @ Wb -> bias[8][L][L] (f32) --------------
__global__ void k_pair_bias(const float* __restrict__ pair,
                            const float* __restrict__ g,
                            const float* __restrict__ b,
                            const float* __restrict__ Wb,
                            float* __restrict__ bias) {
  __shared__ float sWb[PAIR_DIM * NH];
  __shared__ float sg[PAIR_DIM], sb[PAIR_DIM];
  for (int t = threadIdx.x; t < PAIR_DIM * NH; t += 256) sWb[t] = Wb[t];
  for (int t = threadIdx.x; t < PAIR_DIM; t += 256) { sg[t] = g[t]; sb[t] = b[t]; }
  __syncthreads();
  int wave = threadIdx.x >> 5, lane = threadIdx.x & 31;
  size_t flat = (size_t)blockIdx.x * 8 + wave;          // (i,j) flat < L*L
  const float4* pr = (const float4*)(pair + flat * PAIR_DIM);
  float4 x = pr[lane];                                   // 32 lanes * 4 = 128
  float s  = x.x + x.y + x.z + x.w;
  float sq = x.x*x.x + x.y*x.y + x.z*x.z + x.w*x.w;
  s = wave_sum32(s);  sq = wave_sum32(sq);
  float mean = s * (1.0f / 128.0f);
  float var  = sq * (1.0f / 128.0f) - mean * mean;
  float rstd = rsqrtf(var + LN_EPS);
  int d = lane * 4;
  float xa[4] = {x.x, x.y, x.z, x.w};
  float xn[4];
#pragma unroll
  for (int q = 0; q < 4; ++q) xn[q] = (xa[q] - mean) * rstd * sg[d + q] + sb[d + q];
  float myacc = 0.0f;
#pragma unroll
  for (int h = 0; h < 8; ++h) {
    float a = xn[0] * sWb[(d + 0) * 8 + h] + xn[1] * sWb[(d + 1) * 8 + h]
            + xn[2] * sWb[(d + 2) * 8 + h] + xn[3] * sWb[(d + 3) * 8 + h];
    a = wave_sum32(a);
    if (lane == h) myacc = a;
  }
  if (lane < 8) bias[(size_t)lane * LL * LL + flat] = myacc;
}

// ------ Kernel 3: WMMA GEMM  m_bf[8192,256] @ [Wq|Wk|Wv|Wg] -> Q,K,Vt,G ------
__global__ void k_proj_qkvg(const unsigned short* __restrict__ m_bf,
                            const unsigned short* __restrict__ W4t,
                            const float* __restrict__ bg,
                            unsigned short* __restrict__ Qb,
                            unsigned short* __restrict__ Kb,
                            unsigned short* __restrict__ Vt,
                            float* __restrict__ G) {
  int lane = threadIdx.x & 31;
  int wid = blockIdx.x * 4 + (threadIdx.x >> 5);   // 0..32767
  int tmi = wid >> 6, tni = wid & 63;              // 512 x 64 tiles
  int row0 = tmi * 16, n0 = tni * 16;
  int hl = lane >> 4, mr = lane & 15;
  int w = n0 >> 8;                                  // weight segment (uniform)
  int nloc0 = n0 & 255;
  const unsigned short* arow = m_bf + (size_t)(row0 + mr) * 256;
  const unsigned short* bcol = W4t + (size_t)w * 65536 + (size_t)(nloc0 + mr) * 256;
  v8f c = {};
#pragma unroll
  for (int k0 = 0; k0 < 256; k0 += 32) {
    FragAB a, bf;
    a.q[0]  = *(const uint4*)(arow + k0 + hl * 8);
    a.q[1]  = *(const uint4*)(arow + k0 + 16 + hl * 8);
    bf.q[0] = *(const uint4*)(bcol + k0 + hl * 8);
    bf.q[1] = *(const uint4*)(bcol + k0 + 16 + hl * 8);
    c = __builtin_amdgcn_wmma_f32_16x16x32_bf16(false, a.v, false, bf.v,
                                                (short)0, c, false, false);
  }
#pragma unroll
  for (int r = 0; r < 8; ++r) {
    int row = row0 + r + 8 * hl;
    int nl = nloc0 + mr;
    float val = c[r];
    if (w == 0) {
      Qb[(size_t)row * 256 + nl] = f2bf(val * QK_SCALE);
    } else if (w == 1) {
      Kb[(size_t)row * 256 + nl] = f2bf(val);
    } else if (w == 2) {
      int sidx = row >> 10, l = row & 1023;
      int h = nl >> 5, hd = nl & 31;
      Vt[(size_t)((sidx * 8 + h) * 32 + hd) * LL + l] = f2bf(val);
    } else {
      float z = val + bg[nl];
      G[(size_t)row * 256 + nl] = 1.0f / (1.0f + __expf(-z));
    }
  }
}

// ------ Kernel 4: flash attention with pair bias + gate, WMMA bf16 -----------
__global__ void k_attn(const unsigned short* __restrict__ Qb,
                       const unsigned short* __restrict__ Kb,
                       const unsigned short* __restrict__ Vt,
                       const float* __restrict__ bias,
                       const float* __restrict__ G,
                       unsigned short* __restrict__ Ob) {
  __shared__ __align__(16) unsigned short sP[4][16 * 32];
  int wave = threadIdx.x >> 5, lane = threadIdx.x & 31;
  int hl = lane >> 4, mr = lane & 15;
  int wg = blockIdx.x * 4 + wave;                 // 0..4095 = s*512 + h*64 + it
  int sidx = wg >> 9;
  int h    = (wg >> 6) & 7;
  int it   = wg & 63;
  int i0 = it * 16;
  FragAB qa;
  const unsigned short* qrow = Qb + (size_t)(sidx * LL + i0 + mr) * 256 + h * 32;
  qa.q[0] = *(const uint4*)(qrow + hl * 8);
  qa.q[1] = *(const uint4*)(qrow + 16 + hl * 8);
  float run_max[8], run_sum[8];
#pragma unroll
  for (int r = 0; r < 8; ++r) { run_max[r] = -3.0e38f; run_sum[r] = 0.0f; }
  v8f o0 = {}, o1 = {};
  const float* bias_base = bias + (size_t)sidx * LL * LL;
  unsigned short* Pl = sP[wave];
  for (int j0 = 0; j0 < LL; j0 += 32) {
    FragAB kb0, kb1;
    const unsigned short* kr0 = Kb + (size_t)(sidx * LL + j0 + mr) * 256 + h * 32;
    const unsigned short* kr1 = kr0 + 16 * 256;
    kb0.q[0] = *(const uint4*)(kr0 + hl * 8); kb0.q[1] = *(const uint4*)(kr0 + 16 + hl * 8);
    kb1.q[0] = *(const uint4*)(kr1 + hl * 8); kb1.q[1] = *(const uint4*)(kr1 + 16 + hl * 8);
    v8f s0 = {}, s1 = {};
    s0 = __builtin_amdgcn_wmma_f32_16x16x32_bf16(false, qa.v, false, kb0.v,
                                                 (short)0, s0, false, false);
    s1 = __builtin_amdgcn_wmma_f32_16x16x32_bf16(false, qa.v, false, kb1.v,
                                                 (short)0, s1, false, false);
    float p0[8], p1[8];
#pragma unroll
    for (int r = 0; r < 8; ++r) {
      size_t bi = (size_t)(i0 + r + 8 * hl) * LL + j0 + mr;
      p0[r] = s0[r] + bias_base[bi];
      p1[r] = s1[r] + bias_base[bi + 16];
    }
#pragma unroll
    for (int r = 0; r < 8; ++r) {
      float tmax = half_max16(fmaxf(p0[r], p1[r]));
      float nm = fmaxf(run_max[r], tmax);
      float sc = __expf(run_max[r] - nm);
      run_max[r] = nm;
      float e0 = __expf(p0[r] - nm);
      float e1 = __expf(p1[r] - nm);
      run_sum[r] = run_sum[r] * sc + half_sum16(e0 + e1);
      o0[r] *= sc; o1[r] *= sc;
      p0[r] = e0; p1[r] = e1;
    }
    __syncthreads();
#pragma unroll
    for (int r = 0; r < 8; ++r) {
      Pl[(r + 8 * hl) * 32 + mr]      = f2bf(p0[r]);
      Pl[(r + 8 * hl) * 32 + 16 + mr] = f2bf(p1[r]);
    }
    __syncthreads();
    FragAB pa, vb0, vb1;
    pa.q[0] = *(const uint4*)(Pl + mr * 32 + hl * 8);
    pa.q[1] = *(const uint4*)(Pl + mr * 32 + 16 + hl * 8);
    const unsigned short* v0r = Vt + (size_t)((sidx * 8 + h) * 32 + mr) * LL + j0;
    const unsigned short* v1r = v0r + 16 * LL;
    vb0.q[0] = *(const uint4*)(v0r + hl * 8); vb0.q[1] = *(const uint4*)(v0r + 16 + hl * 8);
    vb1.q[0] = *(const uint4*)(v1r + hl * 8); vb1.q[1] = *(const uint4*)(v1r + 16 + hl * 8);
    o0 = __builtin_amdgcn_wmma_f32_16x16x32_bf16(false, pa.v, false, vb0.v,
                                                 (short)0, o0, false, false);
    o1 = __builtin_amdgcn_wmma_f32_16x16x32_bf16(false, pa.v, false, vb1.v,
                                                 (short)0, o1, false, false);
  }
#pragma unroll
  for (int r = 0; r < 8; ++r) {
    int row = i0 + r + 8 * hl;
    float inv = 1.0f / run_sum[r];
    size_t base = (size_t)(sidx * LL + row) * 256 + h * 32;
    float g0 = G[base + mr], g1 = G[base + 16 + mr];
    Ob[base + mr]      = f2bf(o0[r] * inv * g0);
    Ob[base + 16 + mr] = f2bf(o1[r] * inv * g1);
  }
}

// ------ Kernel 5: output projection  Ob[8192,256] @ Wo + bo -> f32 out -------
__global__ void k_out_proj(const unsigned short* __restrict__ Ob,
                           const unsigned short* __restrict__ Wot,
                           const float* __restrict__ bo,
                           float* __restrict__ out) {
  int lane = threadIdx.x & 31;
  int wid = blockIdx.x * 4 + (threadIdx.x >> 5);   // 0..8191 = 512 x 16 tiles
  int tmi = wid >> 4, tni = wid & 15;
  int row0 = tmi * 16, n0 = tni * 16;
  int hl = lane >> 4, mr = lane & 15;
  const unsigned short* arow = Ob  + (size_t)(row0 + mr) * 256;
  const unsigned short* bcol = Wot + (size_t)(n0 + mr) * 256;
  v8f c = {};
#pragma unroll
  for (int k0 = 0; k0 < 256; k0 += 32) {
    FragAB a, bf;
    a.q[0]  = *(const uint4*)(arow + k0 + hl * 8);
    a.q[1]  = *(const uint4*)(arow + k0 + 16 + hl * 8);
    bf.q[0] = *(const uint4*)(bcol + k0 + hl * 8);
    bf.q[1] = *(const uint4*)(bcol + k0 + 16 + hl * 8);
    c = __builtin_amdgcn_wmma_f32_16x16x32_bf16(false, a.v, false, bf.v,
                                                (short)0, c, false, false);
  }
  float bias_n = bo[n0 + mr];
#pragma unroll
  for (int r = 0; r < 8; ++r) {
    out[(size_t)(row0 + r + 8 * hl) * 256 + n0 + mr] = c[r] + bias_n;
  }
}

extern "C" void kernel_launch(void* const* d_in, const int* in_sizes, int n_in,
                              void* d_out, int out_size, void* d_ws, size_t ws_size,
                              hipStream_t stream) {
  (void)in_sizes; (void)n_in; (void)out_size; (void)ws_size;
  const float* msa       = (const float*)d_in[0];
  const float* pair      = (const float*)d_in[1];
  const float* ln_msa_g  = (const float*)d_in[2];
  const float* ln_msa_b  = (const float*)d_in[3];
  const float* ln_pair_g = (const float*)d_in[4];
  const float* ln_pair_b = (const float*)d_in[5];
  const float* Wq        = (const float*)d_in[6];
  const float* Wk        = (const float*)d_in[7];
  const float* Wv        = (const float*)d_in[8];
  const float* Wb        = (const float*)d_in[9];
  const float* Wg        = (const float*)d_in[10];
  const float* bg        = (const float*)d_in[11];
  const float* Wo        = (const float*)d_in[12];
  const float* bo        = (const float*)d_in[13];
  float* out = (float*)d_out;

  char* ws = (char*)d_ws;
  size_t off = 0;
  auto alloc = [&](size_t bytes) -> void* {
    void* p = ws + off;
    off += (bytes + 255) & ~(size_t)255;
    return p;
  };
  unsigned short* m_bf = (unsigned short*)alloc((size_t)SL * 256 * 2);   // 4 MB
  unsigned short* W4t  = (unsigned short*)alloc((size_t)4 * 65536 * 2);  // 512 KB
  unsigned short* Wot  = (unsigned short*)alloc((size_t)65536 * 2);      // 128 KB
  unsigned short* Qb   = (unsigned short*)alloc((size_t)SL * 256 * 2);   // 4 MB
  unsigned short* Kb   = (unsigned short*)alloc((size_t)SL * 256 * 2);   // 4 MB
  unsigned short* Vt   = (unsigned short*)alloc((size_t)SL * 256 * 2);   // 4 MB
  float*          G    = (float*)alloc((size_t)SL * 256 * 4);            // 8 MB
  float*          bias = (float*)alloc((size_t)NH * LL * LL * 4);        // 32 MB
  unsigned short* Ob   = (unsigned short*)alloc((size_t)SL * 256 * 2);   // 4 MB

  k_ln_msa   <<<SL / 8,          256, 0, stream>>>(msa, ln_msa_g, ln_msa_b, m_bf);
  k_wconv    <<<5 * 65536 / 256, 256, 0, stream>>>(Wq, Wk, Wv, Wg, Wo, W4t, Wot);
  k_pair_bias<<<(LL * LL) / 8,   256, 0, stream>>>(pair, ln_pair_g, ln_pair_b, Wb, bias);
  k_proj_qkvg<<<8192,            128, 0, stream>>>(m_bf, W4t, bg, Qb, Kb, Vt, G);
  k_attn     <<<1024,            128, 0, stream>>>(Qb, Kb, Vt, bias, G, Ob);
  k_out_proj <<<2048,            128, 0, stream>>>(Ob, Wot, bo, out);
}
